// VarianceAdaptor_30803505447411
// MI455X (gfx1250) — matrix operand order repaired and checked by reading, not compiled, not run on hardware
//
#include <hip/hip_runtime.h>
#include <hip/hip_bf16.h>

typedef __attribute__((ext_vector_type(16))) _Float16 v16h;
typedef __attribute__((ext_vector_type(8)))  float    v8f;

#define BB   32
#define SS   512
#define HH   384
#define MEL  2048
#define NB   256
#define LNEPS 1e-5f

// ---------------------------------------------------------------------------
// Convert x (f32) * src_mask -> f16
// ---------------------------------------------------------------------------
__global__ __launch_bounds__(256) void va_cvt_x_kernel(
    const float* __restrict__ x, const float* __restrict__ mask,
    _Float16* __restrict__ out, int total)
{
    int i = blockIdx.x * 256 + threadIdx.x;
    if (i >= total) return;
    float m = mask[i / HH];
    out[i] = (_Float16)(x[i] * m);
}

// ---------------------------------------------------------------------------
// Pack conv weights (3*384, 384) f32 -> f16 WMMA B-fragment tiles.
// Tile (kt,nt): 32 lanes x 16 halves contiguous. Lane l: col = l&15,
// Kbase = (l>=16)?16:0, half 2v,2v+1 hold K = Kbase+2v, +1 (v=0..7).
// ---------------------------------------------------------------------------
__global__ __launch_bounds__(256) void va_pack_w_kernel(
    const float* __restrict__ w, _Float16* __restrict__ dst)
{
    int i = blockIdx.x * 256 + threadIdx.x;      // one u32 pair per thread
    const int NTILE = (3 * HH / 32) * (HH / 16); // 36*24 = 864
    if (i >= NTILE * 256) return;
    int tile = i >> 8;
    int within = i & 255;
    int lane = within >> 3;
    int d = within & 7;
    int kt = tile / 24, nt = tile % 24;
    int k0 = kt * 32 + ((lane >= 16) ? 16 : 0) + 2 * d;
    int col = nt * 16 + (lane & 15);
    _Float16* p = dst + (size_t)tile * 512 + lane * 16 + 2 * d;
    p[0] = (_Float16)w[(size_t)k0 * HH + col];
    p[1] = (_Float16)w[(size_t)(k0 + 1) * HH + col];
}

// ---------------------------------------------------------------------------
// Fused predictor conv layer: conv1d(k=3, zero pad) + bias + ReLU + LayerNorm.
// One block = 32 time rows x 384 channels. 8 waves; each wave computes
// 2 M-tiles x 3 N-tiles (B fragments reused across both M-tiles -> halves
// L2 weight traffic vs 16-row blocks).
// If ln_out != null: write LN result as f16 (first conv layer).
// Else: apply gamma/beta, dot with wl, +bl, *mask -> pred (second layer).
// ---------------------------------------------------------------------------
__global__ __launch_bounds__(256) void va_predictor_conv_kernel(
    const _Float16* __restrict__ in,    // (B,T,384) f16
    const _Float16* __restrict__ wpk,   // packed 864 tiles * 512 halves
    const float* __restrict__ bias,     // 384
    const float* __restrict__ gamma,    // 384
    const float* __restrict__ beta,     // 384
    _Float16* __restrict__ ln_out,      // (B,T,384) or null
    const float* __restrict__ wl,       // 384 (unused if ln_out)
    const float* __restrict__ blp,      // 1
    float* __restrict__ pred,           // (B,T)
    const int* __restrict__ mel_len,    // (B) or null
    int T)
{
    __shared__ _Float16 aw[34 * HH];        // rows t0-1 .. t0+32
    __shared__ float hbuf[32 * HH];
    __shared__ float red[2][32][8];
    __shared__ float mrs[2][32];

    const int b   = blockIdx.y;
    const int t0  = blockIdx.x * 32;
    const int tid = threadIdx.x;
    const int lane = tid & 31;
    const int wave = tid >> 5;

    // Stage input window rows t0-1 .. t0+32 (zero pad outside [0,T))
    const _Float16* inb = in + (size_t)b * T * HH;
    for (int i = tid; i < 34 * (HH / 2); i += 256) {
        int row  = i / (HH / 2);
        int col2 = i % (HH / 2);
        int t = t0 - 1 + row;
        unsigned v = 0u;
        if (t >= 0 && t < T)
            v = *(const unsigned*)(inb + (size_t)t * HH + col2 * 2);
        *(unsigned*)(&aw[row * HH + col2 * 2]) = v;
    }
    __syncthreads();

    v8f c00 = {}, c01 = {}, c02 = {};   // M-tile 0, N-tiles 0..2
    v8f c10 = {}, c11 = {}, c12 = {};   // M-tile 1
    const int m  = lane & 15;
    const int kA = (lane >> 4) * 16;

    for (int kappa = 0; kappa < 3; ++kappa) {
        const int row0 = m + kappa;          // window row for M-tile 0
        for (int j = 0; j < 12; ++j) {
            v16h a0, a1;
            unsigned* au0 = (unsigned*)&a0;
            unsigned* au1 = (unsigned*)&a1;
#pragma unroll
            for (int d = 0; d < 8; ++d) {
                int k0 = kA + 2 * d;
                au0[d] = *(const unsigned*)(&aw[row0 * HH + j * 32 + k0]);
                au1[d] = *(const unsigned*)(&aw[(row0 + 16) * HH + j * 32 + k0]);
            }
            int ktile = kappa * 12 + j;
            const _Float16* tb = wpk + ((size_t)(ktile * 24 + wave * 3)) * 512 + lane * 16;
            v16h b0 = *(const v16h*)(tb);
            v16h b1 = *(const v16h*)(tb + 512);
            v16h b2 = *(const v16h*)(tb + 1024);
            c00 = __builtin_amdgcn_wmma_f32_16x16x32_f16(false, a0, false, b0, (short)0, c00, false, false);
            c10 = __builtin_amdgcn_wmma_f32_16x16x32_f16(false, a1, false, b0, (short)0, c10, false, false);
            c01 = __builtin_amdgcn_wmma_f32_16x16x32_f16(false, a0, false, b1, (short)0, c01, false, false);
            c11 = __builtin_amdgcn_wmma_f32_16x16x32_f16(false, a1, false, b1, (short)0, c11, false, false);
            c02 = __builtin_amdgcn_wmma_f32_16x16x32_f16(false, a0, false, b2, (short)0, c02, false, false);
            c12 = __builtin_amdgcn_wmma_f32_16x16x32_f16(false, a1, false, b2, (short)0, c12, false, false);
        }
    }

    // bias + ReLU -> hbuf. C layout: vgpr r -> row r (lanes<16) / r+8 (lanes>=16)
    const int rbase = (lane >> 4) * 8;
    const int col0  = wave * 48 + (lane & 15);
#pragma unroll
    for (int mt = 0; mt < 2; ++mt) {
#pragma unroll
        for (int nt = 0; nt < 3; ++nt) {
            int col = col0 + nt * 16;
            float bsv = bias[col];
            v8f av = (mt == 0) ? ((nt == 0) ? c00 : (nt == 1) ? c01 : c02)
                               : ((nt == 0) ? c10 : (nt == 1) ? c11 : c12);
#pragma unroll
            for (int r = 0; r < 8; ++r) {
                float v = av[r] + bsv;
                hbuf[(mt * 16 + rbase + r) * HH + col] = v > 0.f ? v : 0.f;
            }
        }
    }
    __syncthreads();

    // LayerNorm stats per row (32 rows, 8 partial lanes each)
    {
        int row = tid >> 3, sub = tid & 7;
        float s = 0.f, s2 = 0.f;
        for (int c = sub; c < HH; c += 8) {
            float v = hbuf[row * HH + c];
            s += v; s2 += v * v;
        }
        red[0][row][sub] = s; red[1][row][sub] = s2;
    }
    __syncthreads();
    if (tid < 32) {
        float s = 0.f, s2 = 0.f;
        for (int i = 0; i < 8; ++i) { s += red[0][tid][i]; s2 += red[1][tid][i]; }
        float mean = s * (1.f / HH);
        float var  = s2 * (1.f / HH) - mean * mean;
        mrs[0][tid] = mean;
        mrs[1][tid] = rsqrtf(var + LNEPS);
    }
    __syncthreads();

    if (ln_out) {
        _Float16* ob = ln_out + ((size_t)b * T + t0) * HH;
        for (int i = tid; i < 32 * HH; i += 256) {
            int row = i / HH, col = i % HH;
            float v = (hbuf[i] - mrs[0][row]) * mrs[1][row] * gamma[col] + beta[col];
            ob[i] = (_Float16)v;
        }
    } else {
        // normalize in place, then linear 384->1
        for (int i = tid; i < 32 * HH; i += 256) {
            int row = i / HH, col = i % HH;
            hbuf[i] = (hbuf[i] - mrs[0][row]) * mrs[1][row] * gamma[col] + beta[col];
        }
        __syncthreads();
        {
            int row = tid >> 3, sub = tid & 7;
            float s = 0.f;
            for (int c = sub; c < HH; c += 8) s += hbuf[row * HH + c] * wl[c];
            red[0][row][sub] = s;
        }
        __syncthreads();
        if (tid < 32) {
            float tot = 0.f;
            for (int i = 0; i < 8; ++i) tot += red[0][tid][i];
            float mv = 1.f;
            if (mel_len) mv = ((t0 + tid) < mel_len[b]) ? 1.f : 0.f;
            pred[(size_t)b * T + t0 + tid] = (tot + blp[0]) * mv;
        }
    }
}

// ---------------------------------------------------------------------------
// Length regulate + bucketize + embedding add.
// One block per batch. Writes: out (B,MEL,384), mel_length (float + int),
// masked xe as f16 for the pitch/energy predictors.
// ---------------------------------------------------------------------------
__global__ __launch_bounds__(256) void va_length_regulate_kernel(
    const float* __restrict__ x, const int* __restrict__ dur,
    const float* __restrict__ pitch_t, const float* __restrict__ energy_t,
    const float* __restrict__ pitch_bins, const float* __restrict__ energy_bins,
    const float* __restrict__ pitch_emb, const float* __restrict__ energy_emb,
    float* __restrict__ out, float* __restrict__ mel_len_f,
    int* __restrict__ mel_len_i, _Float16* __restrict__ xe16)
{
    __shared__ int cum[SS];
    __shared__ short idxs[MEL];
    __shared__ short pb[MEL];
    __shared__ short eb[MEL];

    const int b = blockIdx.x;
    const int tid = threadIdx.x;

    if (tid == 0) {
        int s = 0;
        for (int i = 0; i < SS; ++i) { s += dur[(size_t)b * SS + i]; cum[i] = s; }
        mel_len_f[b] = (float)s;
        mel_len_i[b] = s;
    }
    __syncthreads();
    const int mlen = cum[SS - 1];

    for (int t = tid; t < MEL; t += 256) {
        // searchsorted(cum, t, 'right'): first j with cum[j] > t
        int lo = 0, hi = SS;
        while (lo < hi) { int mid = (lo + hi) >> 1; if (cum[mid] <= t) lo = mid + 1; else hi = mid; }
        idxs[t] = (short)(lo < SS - 1 ? lo : SS - 1);
        // searchsorted(bins, v, 'left'): first j with bins[j] >= v
        float pv = pitch_t[(size_t)b * MEL + t];
        lo = 0; hi = NB - 1;
        while (lo < hi) { int mid = (lo + hi) >> 1; if (pitch_bins[mid] < pv) lo = mid + 1; else hi = mid; }
        pb[t] = (short)lo;
        float ev = energy_t[(size_t)b * MEL + t];
        lo = 0; hi = NB - 1;
        while (lo < hi) { int mid = (lo + hi) >> 1; if (energy_bins[mid] < ev) lo = mid + 1; else hi = mid; }
        eb[t] = (short)lo;
    }
    __syncthreads();

    const float* xb = x + (size_t)b * SS * HH;
    float* ob = out + (size_t)b * MEL * HH;
    _Float16* xeb = xe16 + (size_t)b * MEL * HH;
    for (int t = 0; t < MEL; ++t) {
        int src = idxs[t];
        float valid = (t < mlen) ? 1.f : 0.f;
        const float* xr = xb + (size_t)src * HH;
        const float* pe = pitch_emb + (size_t)pb[t] * HH;
        const float* ee = energy_emb + (size_t)eb[t] * HH;
        for (int h = tid; h < HH; h += 256) {
            float xv = xr[h] * valid;
            ob[(size_t)t * HH + h] = xv + pe[h] + ee[h];
            xeb[(size_t)t * HH + h] = (_Float16)xv;
        }
    }
}

// ---------------------------------------------------------------------------
extern "C" void kernel_launch(void* const* d_in, const int* in_sizes, int n_in,
                              void* d_out, int out_size, void* d_ws, size_t ws_size,
                              hipStream_t stream)
{
    (void)in_sizes; (void)n_in; (void)out_size; (void)ws_size;

    const float* x          = (const float*)d_in[0];
    const float* src_mask   = (const float*)d_in[1];
    const int*   dur_t      = (const int*)d_in[2];
    const float* pitch_t    = (const float*)d_in[3];
    const float* energy_t   = (const float*)d_in[4];
    // predictor params: base 5 (dur), 15 (pitch), 25 (energy)
    // order: w1,b1,g1,be1,w2,b2,g2,be2,wl,bl
    const float* pitch_bins  = (const float*)d_in[35];
    const float* energy_bins = (const float*)d_in[36];
    const float* pitch_emb   = (const float*)d_in[37];
    const float* energy_emb  = (const float*)d_in[38];

    float* outp = (float*)d_out;
    const size_t OUT_OFF = 0;
    const size_t ML_OFF  = (size_t)BB * MEL * HH;
    const size_t DUR_OFF = ML_OFF + BB;
    const size_t PIT_OFF = DUR_OFF + (size_t)BB * SS;
    const size_t EN_OFF  = PIT_OFF + (size_t)BB * MEL;

    // workspace layout
    char* ws = (char*)d_ws;
    size_t o = 0;
    _Float16* x16  = (_Float16*)(ws + o); o += (size_t)BB * SS * HH * 2;
    _Float16* xe16 = (_Float16*)(ws + o); o += (size_t)BB * MEL * HH * 2;
    _Float16* h16  = (_Float16*)(ws + o); o += (size_t)BB * MEL * HH * 2;
    const size_t WPK = 864 * 512; // halves per packed conv layer
    _Float16* wpk[6];
    for (int i = 0; i < 6; ++i) { wpk[i] = (_Float16*)(ws + o); o += WPK * 2; }
    int* mli = (int*)(ws + o); o += 256;

    // 1) cast x * mask -> f16
    {
        int total = BB * SS * HH;
        va_cvt_x_kernel<<<(total + 255) / 256, 256, 0, stream>>>(x, src_mask, x16, total);
    }
    // 2) pack 6 conv weight tensors
    {
        const int widx[6] = {5, 9, 15, 19, 25, 29};
        for (int i = 0; i < 6; ++i)
            va_pack_w_kernel<<<864, 256, 0, stream>>>((const float*)d_in[widx[i]], wpk[i]);
    }
    // 3) duration predictor
    {
        dim3 g(SS / 32, BB);
        va_predictor_conv_kernel<<<g, 256, 0, stream>>>(
            x16, wpk[0], (const float*)d_in[6], (const float*)d_in[7], (const float*)d_in[8],
            h16, nullptr, nullptr, nullptr, nullptr, SS);
        va_predictor_conv_kernel<<<g, 256, 0, stream>>>(
            h16, wpk[1], (const float*)d_in[10], (const float*)d_in[11], (const float*)d_in[12],
            nullptr, (const float*)d_in[13], (const float*)d_in[14],
            outp + DUR_OFF, nullptr, SS);
    }
    // 4) length regulate + embeddings + final out
    va_length_regulate_kernel<<<BB, 256, 0, stream>>>(
        x, dur_t, pitch_t, energy_t, pitch_bins, energy_bins,
        pitch_emb, energy_emb, outp + OUT_OFF, outp + ML_OFF, mli, xe16);
    // 5) pitch predictor
    {
        dim3 g(MEL / 32, BB);
        va_predictor_conv_kernel<<<g, 256, 0, stream>>>(
            xe16, wpk[2], (const float*)d_in[16], (const float*)d_in[17], (const float*)d_in[18],
            h16, nullptr, nullptr, nullptr, nullptr, MEL);
        va_predictor_conv_kernel<<<g, 256, 0, stream>>>(
            h16, wpk[3], (const float*)d_in[20], (const float*)d_in[21], (const float*)d_in[22],
            nullptr, (const float*)d_in[23], (const float*)d_in[24],
            outp + PIT_OFF, mli, MEL);
    }
    // 6) energy predictor
    {
        dim3 g(MEL / 32, BB);
        va_predictor_conv_kernel<<<g, 256, 0, stream>>>(
            xe16, wpk[4], (const float*)d_in[26], (const float*)d_in[27], (const float*)d_in[28],
            h16, nullptr, nullptr, nullptr, nullptr, MEL);
        va_predictor_conv_kernel<<<g, 256, 0, stream>>>(
            h16, wpk[5], (const float*)d_in[30], (const float*)d_in[31], (const float*)d_in[32],
            nullptr, (const float*)d_in[33], (const float*)d_in[34],
            outp + EN_OFF, mli, MEL);
    }
}
